// GraphNeuralScheduler_73443940762207
// MI455X (gfx1250) — compile-verified, matrix operand order checked by reading, not run on hardware
//
#include <hip/hip_runtime.h>
#include <hip/hip_fp16.h>

#define N_NODES 50000
#define N_EDGES 800000
#define HIDDEN  128

typedef __attribute__((ext_vector_type(16))) _Float16     v16h;
typedef __attribute__((ext_vector_type(8)))  float        v8f;
typedef __attribute__((ext_vector_type(4)))  unsigned int v4u;

union Frag { v16h v; v4u q[2]; };

__device__ __forceinline__ unsigned int pack2h(float a, float b) {
    union { _Float16 h[2]; unsigned int u; } c;
    c.h[0] = (_Float16)a; c.h[1] = (_Float16)b;
    return c.u;
}
__device__ __forceinline__ __half2 u2h2(unsigned int u) {
    union { unsigned int u; __half2 h; } c; c.u = u; return c.h;
}

// ---- gfx1250 async global->LDS copy (guarded; falls back to sync copy) ----
#if __has_builtin(__builtin_amdgcn_global_load_async_to_lds_b128)
#define HAVE_ASYNC_LDS 1
#else
#define HAVE_ASYNC_LDS 0
#endif

#if HAVE_ASYNC_LDS
typedef int v4i_vs __attribute__((vector_size(16)));
typedef __attribute__((address_space(1))) v4i_vs* g_v4i_p;
typedef __attribute__((address_space(3))) v4i_vs* l_v4i_p;
#endif

__device__ __forceinline__ void copy16_g2lds(const void* gsrc, void* ldst) {
#if HAVE_ASYNC_LDS
    __builtin_amdgcn_global_load_async_to_lds_b128(
        (g_v4i_p)(unsigned long long)gsrc,
        (l_v4i_p)(unsigned int)(size_t)ldst,
        0, 0);
#else
    *(v4u*)ldst = *(const v4u*)gsrc;
#endif
}
__device__ __forceinline__ void wait_async_copies() {
#if HAVE_ASYNC_LDS
#if __has_builtin(__builtin_amdgcn_s_wait_asynccnt)
    __builtin_amdgcn_s_wait_asynccnt(0);
#else
    asm volatile("s_wait_asynccnt 0" ::: "memory");
#endif
#endif
}

// ---------------------------------------------------------------------------
// One-time weight prep: Wt[n][k] = f16(W[k][n])  (transposed, f16 raw bits)
// ---------------------------------------------------------------------------
__global__ __launch_bounds__(256)
void convert_weight_kernel(const float* __restrict__ W, unsigned short* __restrict__ Wt, int K)
{
    const int total = 128 * K;
    for (int i = blockIdx.x * 256 + threadIdx.x; i < total; i += gridDim.x * 256) {
        int n = i & 127;
        int k = i >> 7;
        Wt[(size_t)n * K + k] = (unsigned short)(pack2h(W[(size_t)k * 128 + n], 0.0f) & 0xFFFFu);
    }
}

// ---------------------------------------------------------------------------
// Node encoder: x[M][128] = f16(relu(NF @ W_node + b))   (fp32 A, converted)
// ---------------------------------------------------------------------------
__global__ __launch_bounds__(256)
void gemm_node_kernel(const float* __restrict__ A, const unsigned short* __restrict__ Wt,
                      const float* __restrict__ bias, __half* __restrict__ out)
{
    __shared__ alignas(16) unsigned short ldsA[16 * 128];   // [m][k] f16
    const int tid = threadIdx.x, wave = tid >> 5, lane = tid & 31;
    const int row0 = blockIdx.x * 16;

    for (int i = tid; i < 512; i += 256) {                  // 16 rows x 32 float4s
        int r = i >> 5, kq = i & 31;
        float4 v = *(const float4*)&A[(size_t)(row0 + r) * 128 + kq * 4];
        uint2 p; p.x = pack2h(v.x, v.y); p.y = pack2h(v.z, v.w);
        *(uint2*)&ldsA[r * 128 + kq * 4] = p;
    }
    __syncthreads();

    const int nloc = lane & 15;
    const int koff = (lane < 16) ? 0 : 8;
    const int ncol = wave * 16 + nloc;
    const unsigned short* wrow = Wt + (size_t)ncol * 128;

    v8f acc = {};
#pragma unroll
    for (int k0 = 0; k0 < 128; k0 += 32) {
        Frag af, bf;
        af.q[0] = *(const v4u*)&ldsA[nloc * 128 + k0 + koff];
        af.q[1] = *(const v4u*)&ldsA[nloc * 128 + k0 + koff + 16];
        bf.q[0] = *(const v4u*)&wrow[k0 + koff];
        bf.q[1] = *(const v4u*)&wrow[k0 + koff + 16];
        acc = __builtin_amdgcn_wmma_f32_16x16x32_f16(
            false, af.v, false, bf.v, (short)0, acc, false, false);
    }

    const float b = bias[ncol];
    const int mbase = (lane < 16) ? 0 : 8;
#pragma unroll
    for (int j = 0; j < 8; ++j) {                            // branch-free b16 stores
        float v = acc[j] + b; v = v > 0.0f ? v : 0.0f;
        out[(size_t)(row0 + mbase + j) * 128 + ncol] = __float2half(v);
    }
}

// ---------------------------------------------------------------------------
// Conv GEMM: x[M][128] = f16(relu([m_x | m_e] @ W + b)); A-tile staged via
// GLOBAL_LOAD_ASYNC_TO_LDS_B128. K=256 in 8 WMMA steps.
// ---------------------------------------------------------------------------
__global__ __launch_bounds__(256)
void gemm_conv_kernel(const __half* __restrict__ A1, const __half* __restrict__ A2,
                      const unsigned short* __restrict__ Wt,
                      const float* __restrict__ bias, __half* __restrict__ out)
{
    __shared__ alignas(16) unsigned short ldsA[16 * 256];   // [m][k0..255] f16
    const int tid = threadIdx.x, wave = tid >> 5, lane = tid & 31;
    const int row0 = blockIdx.x * 16;

    for (int i = tid; i < 512; i += 256) {                  // 16 rows x 32 x 16B
        int r = i >> 5, c = i & 31;
        const __half* src = (c < 16) ? (A1 + (size_t)(row0 + r) * 128 + c * 8)
                                     : (A2 + (size_t)(row0 + r) * 128 + (c - 16) * 8);
        copy16_g2lds(src, &ldsA[r * 256 + c * 8]);
    }
    wait_async_copies();
    __syncthreads();

    const int nloc = lane & 15;
    const int koff = (lane < 16) ? 0 : 8;
    const int ncol = wave * 16 + nloc;
    const unsigned short* wrow = Wt + (size_t)ncol * 256;

    v8f acc = {};
#pragma unroll
    for (int k0 = 0; k0 < 256; k0 += 32) {
        Frag af, bf;
        af.q[0] = *(const v4u*)&ldsA[nloc * 256 + k0 + koff];
        af.q[1] = *(const v4u*)&ldsA[nloc * 256 + k0 + koff + 16];
        bf.q[0] = *(const v4u*)&wrow[k0 + koff];
        bf.q[1] = *(const v4u*)&wrow[k0 + koff + 16];
        acc = __builtin_amdgcn_wmma_f32_16x16x32_f16(
            false, af.v, false, bf.v, (short)0, acc, false, false);
    }

    const float b = bias[ncol];
    const int mbase = (lane < 16) ? 0 : 8;
#pragma unroll
    for (int j = 0; j < 8; ++j) {                            // branch-free b16 stores
        float v = acc[j] + b; v = v > 0.0f ? v : 0.0f;
        out[(size_t)(row0 + mbase + j) * 128 + ncol] = __float2half(v);
    }
}

// ---------------------------------------------------------------------------
// Edge encoder fused with scatter: m_e[dst[e]] += relu(EF[e] @ W_edge + b)
// Block = 64 edges; one 16x16x32 WMMA per tile; branch-free pk_add_f16 atomics:
// lane pair exchanges values, even lane takes rows 0-3, odd lane rows 4-7.
// ---------------------------------------------------------------------------
__global__ __launch_bounds__(256)
void edge_encode_scatter_kernel(const float* __restrict__ EF,
                                const unsigned short* __restrict__ WtE,
                                const float* __restrict__ bias,
                                const long long* __restrict__ dst,
                                __half* __restrict__ m_e)
{
    __shared__ alignas(16) unsigned short ldsE[64 * 32];    // [edge][k] f16
    __shared__ long long ldsD[64];                          // dst indices
    const int tid = threadIdx.x, wave = tid >> 5, lane = tid & 31;
    const int e0 = blockIdx.x * 64;

    for (int i = tid; i < 512; i += 256) {                  // 64 edges x 8 float4s
        int m = i >> 3, kq = i & 7;
        float4 v = *(const float4*)&EF[(size_t)(e0 + m) * 32 + kq * 4];
        uint2 p; p.x = pack2h(v.x, v.y); p.y = pack2h(v.z, v.w);
        *(uint2*)&ldsE[m * 32 + kq * 4] = p;
    }
    if (tid < 64) ldsD[tid] = dst[e0 + tid];
    __syncthreads();

    const int nloc = lane & 15;
    const int koff = (lane < 16) ? 0 : 8;
    const int ncol = wave * 16 + nloc;
    const float b = bias[ncol];
    const int mbase = (lane < 16) ? 0 : 8;
    const bool odd = (lane & 1) != 0;
    const int colbase = ncol & ~1;

    Frag bf;
    bf.q[0] = *(const v4u*)&WtE[(size_t)ncol * 32 + koff];
    bf.q[1] = *(const v4u*)&WtE[(size_t)ncol * 32 + koff + 16];

#pragma unroll
    for (int t = 0; t < 4; ++t) {
        Frag af;
        af.q[0] = *(const v4u*)&ldsE[(t * 16 + nloc) * 32 + koff];
        af.q[1] = *(const v4u*)&ldsE[(t * 16 + nloc) * 32 + koff + 16];
        v8f acc = {};
        acc = __builtin_amdgcn_wmma_f32_16x16x32_f16(
            false, af.v, false, bf.v, (short)0, acc, false, false);

        float va[8], vq[8];
#pragma unroll
        for (int j = 0; j < 8; ++j) {
            float v = acc[j] + b; v = v > 0.0f ? v : 0.0f;
            va[j] = v;
            vq[j] = __shfl_xor(v, 1, 32);                   // partner's value (col^1)
        }
#pragma unroll
        for (int jj = 0; jj < 4; ++jj) {
            // even lane -> row jj, odd lane -> row jj+4 (cndmask selects, no branch)
            float own  = odd ? va[jj + 4] : va[jj];
            float par  = odd ? vq[jj + 4] : vq[jj];
            float lo   = odd ? par : own;                   // value at colbase
            float hi   = odd ? own : par;                   // value at colbase+1
            int   row  = mbase + jj + (odd ? 4 : 0);
            long long d = ldsD[t * 16 + row];
            atomicAdd((__half2*)&m_e[(size_t)d * 128 + colbase],
                      __halves2half2(__float2half(lo), __float2half(hi)));
        }
    }
}

// m_x[dst[e]] += x[src[e]] : one wave per edge, f16 path
__global__ __launch_bounds__(256)
void gather_scatter_kernel(const __half* __restrict__ x,
                           const long long* __restrict__ src,
                           const long long* __restrict__ dst,
                           __half* __restrict__ m_x)
{
    const int wave = threadIdx.x >> 5;
    const int lane = threadIdx.x & 31;
    const int e = blockIdx.x * 8 + wave;
    long long s = src[e];
    long long d = dst[e];
    uint2 v = ((const uint2*)(x + (size_t)s * 128))[lane];  // 4 f16 per lane
    __half2* o = (__half2*)(m_x + (size_t)d * 128) + lane * 2;
    atomicAdd(o + 0, u2h2(v.x));
    atomicAdd(o + 1, u2h2(v.y));
}

__global__ __launch_bounds__(256)
void zero_kernel(unsigned int* __restrict__ p, size_t n)
{
    size_t i = (size_t)blockIdx.x * blockDim.x + threadIdx.x;
    size_t stride = (size_t)gridDim.x * blockDim.x;
    for (; i < n; i += stride) p[i] = 0u;
}

// scores[node] = x[node] . W_out + b_out : wave per node, shuffle reduce
__global__ __launch_bounds__(256)
void score_kernel(const __half* __restrict__ x, const float* __restrict__ Wout,
                  const float* __restrict__ bout, float* __restrict__ scores)
{
    const int wave = threadIdx.x >> 5;
    const int lane = threadIdx.x & 31;
    const int node = blockIdx.x * 8 + wave;
    uint2 a = ((const uint2*)(x + (size_t)node * 128))[lane];
    float4 w = ((const float4*)Wout)[lane];
    float2 a0 = __half22float2(u2h2(a.x));
    float2 a1 = __half22float2(u2h2(a.y));
    float p = a0.x * w.x + a0.y * w.y + a1.x * w.z + a1.y * w.w;
#pragma unroll
    for (int off = 16; off > 0; off >>= 1)
        p += __shfl_down(p, off, 32);
    if (lane == 0) scores[node] = p + bout[0];
}

// Single-block two-pass softmax (in place)
__global__ __launch_bounds__(1024)
void softmax_kernel(const float* __restrict__ scores, float* __restrict__ out, int M)
{
    __shared__ float red[32];
    const int tid = threadIdx.x;

    float mx = -3.402823e38f;
    for (int i = tid; i < M; i += 1024) mx = fmaxf(mx, scores[i]);
#pragma unroll
    for (int off = 16; off > 0; off >>= 1) mx = fmaxf(mx, __shfl_xor(mx, off, 32));
    if ((tid & 31) == 0) red[tid >> 5] = mx;
    __syncthreads();
    if (tid < 32) {
        float v = red[tid];
#pragma unroll
        for (int off = 16; off > 0; off >>= 1) v = fmaxf(v, __shfl_xor(v, off, 32));
        red[tid] = v;
    }
    __syncthreads();
    mx = red[0];
    __syncthreads();

    float sum = 0.0f;
    for (int i = tid; i < M; i += 1024) sum += __expf(scores[i] - mx);
#pragma unroll
    for (int off = 16; off > 0; off >>= 1) sum += __shfl_xor(sum, off, 32);
    if ((tid & 31) == 0) red[tid >> 5] = sum;
    __syncthreads();
    if (tid < 32) {
        float v = red[tid];
#pragma unroll
        for (int off = 16; off > 0; off >>= 1) v += __shfl_xor(v, off, 32);
        red[tid] = v;
    }
    __syncthreads();
    const float inv = 1.0f / red[0];
    for (int i = tid; i < M; i += 1024) out[i] = __expf(scores[i] - mx) * inv;
}

extern "C" void kernel_launch(void* const* d_in, const int* in_sizes, int n_in,
                              void* d_out, int out_size, void* d_ws, size_t ws_size,
                              hipStream_t stream) {
    const float*     node_features = (const float*)d_in[0];
    const float*     edge_features = (const float*)d_in[1];
    const float*     W_node = (const float*)d_in[2];
    const float*     b_node = (const float*)d_in[3];
    const float*     W_edge = (const float*)d_in[4];
    const float*     b_edge = (const float*)d_in[5];
    const float*     W1     = (const float*)d_in[6];
    const float*     b1     = (const float*)d_in[7];
    const float*     W2     = (const float*)d_in[8];
    const float*     b2     = (const float*)d_in[9];
    const float*     W_out  = (const float*)d_in[10];
    const float*     b_out  = (const float*)d_in[11];
    const long long* eidx   = (const long long*)d_in[12];
    const long long* src = eidx;
    const long long* dst = eidx + N_EDGES;

    const size_t NH = (size_t)N_NODES * HIDDEN;
    __half* x   = (__half*)d_ws;                       // [N,128] f16 activations
    __half* m_e = x + NH;                              // [N,128] f16 edge messages
    __half* m_x = m_e + NH;                            // [N,128] f16 node messages
    unsigned short* WtN = (unsigned short*)(m_x + NH); // [128][128] f16^T
    unsigned short* Wt1 = WtN + 128 * 128;             // [128][256] f16^T
    unsigned short* Wt2 = Wt1 + 128 * 256;             // [128][256] f16^T
    unsigned short* WtE = Wt2 + 128 * 256;             // [128][32]  f16^T
    float* scores = (float*)d_out;

    // One-time weight transpose+convert to f16 (L2-resident tables)
    convert_weight_kernel<<<64,  256, 0, stream>>>(W_node, WtN, 128);
    convert_weight_kernel<<<128, 256, 0, stream>>>(W1,     Wt1, 256);
    convert_weight_kernel<<<128, 256, 0, stream>>>(W2,     Wt2, 256);
    convert_weight_kernel<<<16,  256, 0, stream>>>(W_edge, WtE, 32);

    // zero m_e and m_x (contiguous, 2*NH halfs = NH dwords)
    zero_kernel<<<2048, 256, 0, stream>>>((unsigned int*)m_e, NH);

    // x = relu(NF @ W_node + b_node)
    gemm_node_kernel<<<N_NODES / 16, 256, 0, stream>>>(node_features, WtN, b_node, x);

    // m_e = segment_sum(relu(EF @ W_edge + b_edge), dst)  [layer-invariant: once]
    edge_encode_scatter_kernel<<<N_EDGES / 64, 256, 0, stream>>>(
        edge_features, WtE, b_edge, dst, m_e);

    // conv layer 1
    gather_scatter_kernel<<<N_EDGES / 8, 256, 0, stream>>>(x, src, dst, m_x);
    gemm_conv_kernel<<<N_NODES / 16, 256, 0, stream>>>(m_x, m_e, Wt1, b1, x);

    // conv layer 2
    zero_kernel<<<2048, 256, 0, stream>>>((unsigned int*)m_x, NH / 2);
    gather_scatter_kernel<<<N_EDGES / 8, 256, 0, stream>>>(x, src, dst, m_x);
    gemm_conv_kernel<<<N_NODES / 16, 256, 0, stream>>>(m_x, m_e, Wt2, b2, x);

    // output head + softmax
    score_kernel<<<N_NODES / 8, 256, 0, stream>>>(x, W_out, b_out, scores);
    softmax_kernel<<<1, 1024, 0, stream>>>(scores, (float*)d_out, N_NODES);
}
